// NMP_Duvenaud_38998303048176
// MI455X (gfx1250) — compile-verified, matrix operand order checked by reading, not run on hardware
//
#include <hip/hip_runtime.h>
#include <hip/hip_bf16.h>
#include <math.h>

// ---------------------------------------------------------------------------
// Duvenaud NMP forward on gfx1250 (MI455X), wave32 + WMMA f16->f32.
//   B=128 graphs, N=128 nodes, F=64 node feats, E=8 edge feats, 12 deg buckets
// HBM-bound on the edge tensor e (134 MB over two layers); all GEMMs run on
// v_wmma_f32_16x16x32_f16 (adjacency is binary -> exact in f16), edge traffic
// is fetched as fully-coalesced 128B bursts and pre-warmed into L2 with
// global_prefetch so it overlaps the WMMA phase.
// ---------------------------------------------------------------------------

#define NN   128
#define FV   64
#define FE   8
#define FIN  72   // FV + FE

typedef __attribute__((ext_vector_type(16))) _Float16 v16h;
typedef __attribute__((ext_vector_type(8)))  _Float16 v8h;
typedef __attribute__((ext_vector_type(4)))  _Float16 v4h;
typedef __attribute__((ext_vector_type(8)))  float    v8f;

// Load one lane's A/B fragment for V_WMMA_F32_16X16X32_F16 from an LDS array
// stored [row][k] row-major with stride `stride` halves.
// ISA layout (16-bit 16x32 A; B symmetric with N in place of M):
//   lanes 0-15 : row = base+lane,    K = kb+{0..7}  and kb+16+{0..7}
//   lanes 16-31: row = base+lane-16, K = kb+{8..15} and kb+16+{8..15}
// -> two contiguous 16-byte chunks per lane (ds_load_b128 each).
__device__ inline v16h load_frag(const _Float16* base_ptr, int row, int stride,
                                 int kb, int lane) {
  union { v16h v; v8h h[2]; } u;
  const int half = (lane >> 4) & 1;
  const _Float16* p0 = base_ptr + row * stride + kb + half * 8;
  u.h[0] = *(const v8h*)(p0);
  u.h[1] = *(const v8h*)(p0 + 16);
  return u.v;
}

// Store a 16x16 f32 D fragment: VGPR i, lanes 0-15 -> M=i, lanes 16-31 -> M=i+8,
// N = lane & 15.
__device__ inline void store_frag(float* dst, int mrow, int ncol, int stride,
                                  int lane, v8f c) {
  const int half = (lane >> 4) & 1;
  const int n = lane & 15;
#pragma unroll
  for (int i = 0; i < 8; ++i)
    dst[(mrow + half * 8 + i) * stride + ncol + n] = c[i];
}

__device__ inline float sigmoidf_(float x) { return 1.0f / (1.0f + __expf(-x)); }

// ---------------------------------------------------------------------------
// One Duvenaud layer: message pass + degree-bucketed update.
// grid = 2*B blocks; block handles 64 node rows of one graph. 256 thr = 8 waves.
// ---------------------------------------------------------------------------
__global__ __launch_bounds__(256)
void duv_layer_kernel(const float* __restrict__ g,
                      const float* __restrict__ h,
                      const float* __restrict__ e,
                      const float* __restrict__ Hd,   // [12][72][64]
                      float* __restrict__ h_out) {
  __shared__ __align__(16) _Float16 gh[64 * NN];   // adjacency chunk, f16 (exact)
  __shared__ __align__(16) _Float16 hT[FV * NN];   // h transposed [f][w]
  __shared__ float sm[64 * FIN];                   // messages for chunk
  __shared__ int   sidx[64];                       // degree bucket (-1 invalid)

  const int b     = blockIdx.x >> 1;
  const int vbase = (blockIdx.x & 1) * 64;
  const int t     = threadIdx.x;
  const int lane  = t & 31;
  const int wave  = t >> 5;

  const float* gB = g + (size_t)b * NN * NN;
  const float* hB = h + (size_t)b * NN * FV;
  const float* eB = e + (size_t)b * NN * NN * FE;

  // ---- stage adjacency rows (f16, b128 global loads / b64 LDS stores) ----
  for (int i = t; i < (64 * NN) / 4; i += 256) {
    const int v = (i * 4) >> 7, w = (i * 4) & 127;
    const float4 g4 = *(const float4*)(gB + (vbase + v) * NN + w);
    v4h hv;
    hv[0] = (_Float16)g4.x; hv[1] = (_Float16)g4.y;
    hv[2] = (_Float16)g4.z; hv[3] = (_Float16)g4.w;
    *(v4h*)(gh + v * NN + w) = hv;
  }
  // ---- stage h transposed: coalesced b128 reads, scattered 16-bit LDS writes
  for (int i = t; i < (NN * FV) / 4; i += 256) {
    const int w = (i * 4) >> 6, f = (i * 4) & 63;
    const float4 h4 = *(const float4*)(hB + w * FV + f);
    hT[(f + 0) * NN + w] = (_Float16)h4.x;
    hT[(f + 1) * NN + w] = (_Float16)h4.y;
    hT[(f + 2) * NN + w] = (_Float16)h4.z;
    hT[(f + 3) * NN + w] = (_Float16)h4.w;
  }
  // ---- warm L2 with this block's 256KB slice of e while WMMA runs ----
  {
    const float* ebase = eB + (size_t)vbase * NN * FE;
#pragma unroll 2
    for (int i = t; i < (64 * NN * FE) / 32; i += 256)
      __builtin_prefetch(ebase + (size_t)i * 32, 0, 0);
  }
  __syncthreads();

  // ---- WMMA GEMM: sm[:, :64] = g_chunk (64x128) @ h (128x64), f32 acc ----
  {
    const int mrow = (wave >> 1) * 16;               // 4 row strips
    const int mr   = mrow + (lane & 15);
#pragma unroll
    for (int cpair = 0; cpair < 2; ++cpair) {
      const int ncol = ((wave & 1) * 2 + cpair) * 16;
      const int nr   = ncol + (lane & 15);
      v8f acc = {};
#pragma unroll
      for (int kb = 0; kb < NN; kb += 32) {
        v16h a  = load_frag(gh, mr, NN, kb, lane);
        v16h bm = load_frag(hT, nr, NN, kb, lane);
        acc = __builtin_amdgcn_wmma_f32_16x16x32_f16(
            false, a, false, bm, (short)0, acc, false, false);
      }
      store_frag(sm, mrow, ncol, FIN, lane, acc);
    }
  }

  // ---- degrees + bucket indices (jnp.searchsorted over DEGS=1..12) ----
  if (t < 64) {
    float d = 0.f;
#pragma unroll 8
    for (int w = 0; w < NN; ++w) d += (float)gh[t * NN + w];
    const int di = (int)(d + 0.5f);
    sidx[t] = (di >= 1 && di <= 12) ? (di - 1) : -1;
  }

  // ---- edge message: sm[v][64+ec] = sum_w g[v,w]*e[v,w,ec] ----
  // 8 lanes per node; lane j covers w-phase (j>>1) and channels (j&1)*4..+3.
  // Per group iteration: 8 lanes x float4 = 128B fully contiguous (b128).
  {
    const int j    = t & 7;
    const int v0   = t >> 3;          // 0..31
    const int woff = j >> 1;          // 0..3
    const int ecg  = (j & 1) * 4;     // channel group base
#pragma unroll
    for (int it = 0; it < 2; ++it) {
      const int v = v0 + it * 32;
      const float* ev = eB + (size_t)(vbase + v) * NN * FE;
      float ax = 0.f, ay = 0.f, az = 0.f, aw = 0.f;
#pragma unroll 4
      for (int w = 0; w < NN; w += 4) {
        const int wl = w + woff;
        const float4 e4 = *(const float4*)(ev + wl * FE + ecg);
        const float gv = (float)gh[v * NN + wl];
        ax += gv * e4.x; ay += gv * e4.y;
        az += gv * e4.z; aw += gv * e4.w;
      }
      // reduce over the 4 w-phases (j bits 1..2)
#pragma unroll
      for (int m = 2; m <= 4; m <<= 1) {
        ax += __shfl_xor(ax, m, 32);
        ay += __shfl_xor(ay, m, 32);
        az += __shfl_xor(az, m, 32);
        aw += __shfl_xor(aw, m, 32);
      }
      if (j < 2) {
        sm[v * FIN + FV + ecg + 0] = ax;
        sm[v * FIN + FV + ecg + 1] = ay;
        sm[v * FIN + FV + ecg + 2] = az;
        sm[v * FIN + FV + ecg + 3] = aw;
      }
    }
  }
  __syncthreads();

  // ---- per-node update: h_new = sigmoid(m @ H[bucket]) * valid ----
  {
    const int o = t & 63;
#pragma unroll 2
    for (int pass = 0; pass < 16; ++pass) {
      const int v  = (t >> 6) + pass * 4;
      const int bi = sidx[v];
      float r = 0.f;
      if (bi >= 0) {
        const float* Hm = Hd + (size_t)bi * FIN * FV + o;
        float acc = 0.f;
#pragma unroll 8
        for (int i = 0; i < FIN; ++i)
          acc += sm[v * FIN + i] * Hm[i * FV];
        r = sigmoidf_(acc);
      }
      h_out[((size_t)b * NN + vbase + v) * FV + o] = r;
    }
  }
}

// ---------------------------------------------------------------------------
// Readout: acc[b] = sum_l sum_v softmax(h_l[b] @ R_l)[v] * mask(v)
// grid = B blocks, 256 threads.
// ---------------------------------------------------------------------------
__global__ __launch_bounds__(256)
void readout_kernel(const float* __restrict__ h0,
                    const float* __restrict__ h1,
                    const float* __restrict__ h2,
                    const float* __restrict__ R0,
                    const float* __restrict__ R1,
                    const float* __restrict__ R2,
                    float* __restrict__ acc_out) {
  __shared__ __align__(16) _Float16 Ah[NN * FV];   // 16 KB
  __shared__ __align__(16) _Float16 BT[FV * FV];   // 8 KB (R transposed [n][k])
  __shared__ float sp[NN * FV];                    // 32 KB

  const int b    = blockIdx.x;
  const int t    = threadIdx.x;
  const int lane = t & 31;
  const int wave = t >> 5;

  const float* hs[3] = { h0 + (size_t)b * NN * FV,
                         h1 + (size_t)b * NN * FV,
                         h2 + (size_t)b * NN * FV };
  const float* Rs[3] = { R0, R1, R2 };

  float accR = 0.f;                                // per-output-feature (t<64)

  for (int l = 0; l < 3; ++l) {
    for (int i = t; i < (NN * FV) / 4; i += 256) {
      const float4 h4 = *(const float4*)(hs[l] + i * 4);
      v4h hv;
      hv[0] = (_Float16)h4.x; hv[1] = (_Float16)h4.y;
      hv[2] = (_Float16)h4.z; hv[3] = (_Float16)h4.w;
      *(v4h*)(Ah + i * 4) = hv;
    }
    for (int i = t; i < (FV * FV) / 4; i += 256) {
      const int k = (i * 4) >> 6, n = (i * 4) & 63;
      const float4 r4 = *(const float4*)(Rs[l] + k * FV + n);
      BT[(n + 0) * FV + k] = (_Float16)r4.x;
      BT[(n + 1) * FV + k] = (_Float16)r4.y;
      BT[(n + 2) * FV + k] = (_Float16)r4.z;
      BT[(n + 3) * FV + k] = (_Float16)r4.w;
    }
    __syncthreads();

    // p = h_l (128x64) @ R_l (64x64) via WMMA
    {
      const int mrow = wave * 16;
      const int mr   = mrow + (lane & 15);
#pragma unroll
      for (int ct = 0; ct < 4; ++ct) {
        const int ncol = ct * 16;
        const int nr   = ncol + (lane & 15);
        v8f acc = {};
#pragma unroll
        for (int kb = 0; kb < FV; kb += 32) {
          v16h a  = load_frag(Ah, mr, FV, kb, lane);
          v16h bm = load_frag(BT, nr, FV, kb, lane);
          acc = __builtin_amdgcn_wmma_f32_16x16x32_f16(
              false, a, false, bm, (short)0, acc, false, false);
        }
        store_frag(sp, mrow, ncol, FV, lane, acc);
      }
    }
    __syncthreads();

    // masked softmax per node row, in place
    if (t < NN) {
      float mx = -INFINITY; int nz = 0;
#pragma unroll 8
      for (int o = 0; o < FV; ++o) {
        const float x = sp[t * FV + o];
        nz |= (x != 0.f);
        mx = fmaxf(mx, x);
      }
      float s = 0.f;
#pragma unroll 8
      for (int o = 0; o < FV; ++o) s += __expf(sp[t * FV + o] - mx);
      const float scale = nz ? (1.f / s) : 0.f;
#pragma unroll 8
      for (int o = 0; o < FV; ++o)
        sp[t * FV + o] = __expf(sp[t * FV + o] - mx) * scale;
    }
    __syncthreads();

    // sum over nodes
    if (t < FV) {
      float s = 0.f;
#pragma unroll 8
      for (int v = 0; v < NN; ++v) s += sp[v * FV + t];
      accR += s;
    }
    __syncthreads();   // LDS reuse next l
  }
  if (t < FV) acc_out[b * FV + t] = accR;
}

// ---------------------------------------------------------------------------
// Dense MLP layer: y[128 x OUTD] = act(x[128 x IN] @ W + b). grid = 4 blocks.
// ---------------------------------------------------------------------------
__global__ __launch_bounds__(256)
void dense_kernel(const float* __restrict__ x, const float* __restrict__ W,
                  const float* __restrict__ bias, float* __restrict__ y,
                  int IN, int OUTD, int relu) {
  const int rowbase = blockIdx.x * 32;
  for (int idx = threadIdx.x; idx < 32 * OUTD; idx += 256) {
    const int r = rowbase + idx / OUTD;
    const int c = idx % OUTD;
    float acc = bias[c];
    const float* xr = x + (size_t)r * IN;
    const float* wc = W + c;
    for (int i = 0; i < IN; ++i) acc += xr[i] * wc[(size_t)i * OUTD];
    if (relu) acc = fmaxf(acc, 0.f);
    y[(size_t)r * OUTD + c] = acc;
  }
}

// ---------------------------------------------------------------------------
extern "C" void kernel_launch(void* const* d_in, const int* in_sizes, int n_in,
                              void* d_out, int out_size, void* d_ws, size_t ws_size,
                              hipStream_t stream) {
  (void)in_sizes; (void)n_in; (void)out_size; (void)ws_size;
  const float* g  = (const float*)d_in[0];
  const float* h0 = (const float*)d_in[1];
  const float* e  = (const float*)d_in[2];
  const float* H1 = (const float*)d_in[3];
  const float* H2 = (const float*)d_in[4];
  const float* R0 = (const float*)d_in[5];
  const float* R1 = (const float*)d_in[6];
  const float* R2 = (const float*)d_in[7];
  const float* W0 = (const float*)d_in[8];
  const float* W1 = (const float*)d_in[9];
  const float* W2 = (const float*)d_in[10];
  const float* W3 = (const float*)d_in[11];
  const float* b0 = (const float*)d_in[12];
  const float* b1 = (const float*)d_in[13];
  const float* b2 = (const float*)d_in[14];
  const float* b3 = (const float*)d_in[15];

  float* ws  = (float*)d_ws;
  float* h1  = ws;                          // 128*128*64
  float* h2  = h1 + 128 * 128 * 64;         // 128*128*64
  float* acc = h2 + 128 * 128 * 64;         // 128*64
  float* x1  = acc + 128 * 64;              // 128*128
  float* x2  = x1 + 128 * 128;              // 128*256
  float* x3  = x2 + 128 * 256;              // 128*128

  duv_layer_kernel<<<256, 256, 0, stream>>>(g, h0, e, H1, h1);
  duv_layer_kernel<<<256, 256, 0, stream>>>(g, h1, e, H2, h2);
  readout_kernel<<<128, 256, 0, stream>>>(h0, h1, h2, R0, R1, R2, acc);
  dense_kernel<<<4, 256, 0, stream>>>(acc, W0, b0, x1, 64, 128, 1);
  dense_kernel<<<4, 256, 0, stream>>>(x1, W1, b1, x2, 128, 256, 1);
  dense_kernel<<<4, 256, 0, stream>>>(x2, W2, b2, x3, 256, 128, 1);
  dense_kernel<<<4, 256, 0, stream>>>(x3, W3, b3, (float*)d_out, 128, 12, 0);
}
